// DTILJPredictor_17231408792052
// MI455X (gfx1250) — matrix operand order repaired and checked by reading, not compiled, not run on hardware
//
#include <hip/hip_runtime.h>
#include <cstdint>

// Problem constants (match reference)
#define BB   4
#define N1_  128
#define N2_  512
#define DD   128
#define LL   3
#define FEATK 56
#define FEATP 64   // FEAT zero-padded to a multiple of 32 for WMMA K
#define HID_ 128

typedef __attribute__((ext_vector_type(16))) __bf16 v16bf;
typedef __attribute__((ext_vector_type(8)))  float  v8f;

enum { ACT_NONE = 0, ACT_RELU = 1 };

// ---------------------------------------------------------------------------
// gfx1250 async global->LDS copy (ASYNCcnt path), with safe fallback.
// ---------------------------------------------------------------------------
typedef __attribute__((address_space(1))) int gint_t;
typedef __attribute__((address_space(3))) int lint_t;

__device__ __forceinline__ void async_copy_b32(const float* g, float* l) {
#if defined(__gfx1250__) && __has_builtin(__builtin_amdgcn_global_load_async_to_lds_b32)
  __builtin_amdgcn_global_load_async_to_lds_b32((gint_t*)g, (lint_t*)l, 0, 0);
#else
  *l = *g;
#endif
}

__device__ __forceinline__ void wait_async0() {
#if defined(__gfx1250__)
#if __has_builtin(__builtin_amdgcn_s_wait_asynccnt)
  __builtin_amdgcn_s_wait_asynccnt(0);
#else
  asm volatile("s_wait_asynccnt 0" ::: "memory");
#endif
#endif
}

__device__ __forceinline__ float sigmoidf_(float x) {
  return 1.f / (1.f + __expf(-x));
}

// Pack 16 bf16 fragment elements from two contiguous 8-float runs
// (elements 0..7 <- p0[0..7], elements 8..15 <- p1[0..7]); b128 loads.
__device__ __forceinline__ v16bf cvt16(const float* p0, const float* p1) {
  const float4 x0 = ((const float4*)p0)[0];
  const float4 x1 = ((const float4*)p0)[1];
  const float4 y0 = ((const float4*)p1)[0];
  const float4 y1 = ((const float4*)p1)[1];
  v16bf r;
  r[0]  = (__bf16)x0.x; r[1]  = (__bf16)x0.y; r[2]  = (__bf16)x0.z; r[3]  = (__bf16)x0.w;
  r[4]  = (__bf16)x1.x; r[5]  = (__bf16)x1.y; r[6]  = (__bf16)x1.z; r[7]  = (__bf16)x1.w;
  r[8]  = (__bf16)y0.x; r[9]  = (__bf16)y0.y; r[10] = (__bf16)y0.z; r[11] = (__bf16)y0.w;
  r[12] = (__bf16)y1.x; r[13] = (__bf16)y1.y; r[14] = (__bf16)y1.z; r[15] = (__bf16)y1.w;
  return r;
}

// ---------------------------------------------------------------------------
// bf16 WMMA GEMM with B pre-transposed:
//   C[b] = act( A[b] @ Bt[b]^T (+bias) (+C[b] if accum) )
// A:  [M,K] row-major.   Bt: [N,K] row-major (row n = column n of B).
// PRECONDITIONS: K % 32 == 0, M % 64 == 0 or tile-covered, N % 64 == 0
// (call sites use M,N in {128,512,2048} and K in {64,128,256,512}).
// Each wave computes a 32x32 tile (2x2 micro-tiles, 4 accumulators):
// per K-step 16x global_load_b128 feed 4x v_wmma -> 4 loads per WMMA.
// Block = 4 waves in a 2x2 arrangement covering 64x64 of C.
// ---------------------------------------------------------------------------
__global__ void wmma_gemm_kernel(const float* __restrict__ A,
                                 const float* __restrict__ Bt,
                                 const float* __restrict__ bias,
                                 float* __restrict__ C,
                                 int M, int N, int K,
                                 long long sA, long long sB, long long sC,
                                 int accum, int act) {
  const int lane = threadIdx.x & 31;
  const int wave = threadIdx.x >> 5;
  const int row0 = blockIdx.y * 64 + (wave >> 1) * 32;
  const int col0 = blockIdx.x * 64 + (wave & 1) * 32;
  if (col0 >= N || row0 >= M) return;  // wave-uniform
  const float* Ab = A + (long long)blockIdx.z * sA;
  const float* Bb = Bt + (long long)blockIdx.z * sB;
  float* Cb = C + (long long)blockIdx.z * sC;

  // ISA 7.12.2 lane roles (wave32)
  const int m  = lane & 15;             // A row within micro-tile
  const int kh = (lane >> 4) * 8;       // A K-half offset
  const int nn = lane & 15;             // B column within micro-tile
  const int kb = (lane >> 4) * 8;       // B K-half offset
  const float* arow0 = Ab + (long long)(row0 + m) * K;
  const float* arow1 = arow0 + 16LL * K;
  const float* brow0 = Bb + (long long)(col0 + nn) * K;
  const float* brow1 = brow0 + 16LL * K;

  v8f a00 = {}, a01 = {}, a10 = {}, a11 = {};
#pragma unroll 2
  for (int k0 = 0; k0 < K; k0 += 32) {
    const v16bf af0 = cvt16(arow0 + k0 + kh, arow0 + k0 + kh + 16);
    const v16bf af1 = cvt16(arow1 + k0 + kh, arow1 + k0 + kh + 16);
    const v16bf bf0 = cvt16(brow0 + k0 + kb, brow0 + k0 + kb + 16);
    const v16bf bf1 = cvt16(brow1 + k0 + kb, brow1 + k0 + kb + 16);
    a00 = __builtin_amdgcn_wmma_f32_16x16x32_bf16(false, af0, false, bf0, (short)0, a00, false, false);
    a01 = __builtin_amdgcn_wmma_f32_16x16x32_bf16(false, af0, false, bf1, (short)0, a01, false, false);
    a10 = __builtin_amdgcn_wmma_f32_16x16x32_bf16(false, af1, false, bf0, (short)0, a10, false, false);
    a11 = __builtin_amdgcn_wmma_f32_16x16x32_bf16(false, af1, false, bf1, (short)0, a11, false, false);
  }

  // ---- store: f32 C/D layout: lane n-col, 8 rows per lane (m, m+8 halves) ----
  const int mb = 8 * (lane >> 4);
  const float bv0 = bias ? bias[col0 + nn] : 0.f;
  const float bv1 = bias ? bias[col0 + 16 + nn] : 0.f;
#pragma unroll
  for (int q = 0; q < 4; ++q) {
    const v8f acc = (q == 0) ? a00 : (q == 1) ? a01 : (q == 2) ? a10 : a11;
    const int rbase = row0 + ((q >> 1) ? 16 : 0) + mb;
    const int c = col0 + ((q & 1) ? 16 : 0) + nn;
    const float bv = (q & 1) ? bv1 : bv0;
#pragma unroll
    for (int r = 0; r < 8; ++r) {
      const long long idx = (long long)(rbase + r) * N + c;
      float v = acc[r] + bv;
      if (accum) v += Cb[idx];
      if (act == ACT_RELU) v = fmaxf(v, 0.f);
      Cb[idx] = v;
    }
  }
}

// ---------------------------------------------------------------------------
// Batched 16x16-tiled transpose: dst[b][c*R + r] = src[b][r*C + c].
// R % 16 == 0, C % 16 == 0. grid = (C/16, R/16, batch), block = 256.
// ---------------------------------------------------------------------------
__global__ void transpose16_kernel(const float* __restrict__ src,
                                   float* __restrict__ dst,
                                   int R, int C, long long sS, long long sD) {
  __shared__ float tile[16][17];
  const float* s = src + (long long)blockIdx.z * sS;
  float* d = dst + (long long)blockIdx.z * sD;
  const int r0 = blockIdx.y * 16, c0 = blockIdx.x * 16;
  const int tx = threadIdx.x & 15, ty = threadIdx.x >> 4;
  tile[ty][tx] = s[(long long)(r0 + ty) * C + c0 + tx];
  __syncthreads();
  d[(long long)(c0 + ty) * R + r0 + tx] = tile[tx][ty];
}

// ---------------------------------------------------------------------------
// Zero-padded 2D copy: dst[drows,dcols] <- src[srows,scols] (zero fill)
// ---------------------------------------------------------------------------
__global__ void pad2d_kernel(const float* __restrict__ src, float* __restrict__ dst,
                             int srows, int scols, int drows, int dcols) {
  const int t = blockIdx.x * blockDim.x + threadIdx.x;
  if (t >= drows * dcols) return;
  const int r = t / dcols, c = t % dcols;
  dst[t] = (r < srows && c < scols) ? src[r * scols + c] : 0.f;
}

// ---------------------------------------------------------------------------
// GAT attention: e_sym = e + e^T, mask by adj, softmax, * adj  -> att
// grid = (N, B), block = 256
// ---------------------------------------------------------------------------
__global__ void att_softmax_kernel(const float* __restrict__ e,
                                   const float* __restrict__ adj,
                                   float* __restrict__ att, int N) {
  const int b = blockIdx.y, i = blockIdx.x, tid = threadIdx.x;
  const float* eb = e + (long long)b * N * N;
  const float* ab = adj + (long long)b * N * N;
  float* ob = att + (long long)b * N * N;
  __shared__ float red[256];

  float mx = -3.4e38f;
  for (int j = tid; j < N; j += 256) {
    const float val = eb[(long long)i * N + j] + eb[(long long)j * N + i];
    const float a = ab[(long long)i * N + j];
    mx = fmaxf(mx, (a > 0.f) ? val : -9e15f);
  }
  red[tid] = mx; __syncthreads();
  for (int s = 128; s > 0; s >>= 1) {
    if (tid < s) red[tid] = fmaxf(red[tid], red[tid + s]);
    __syncthreads();
  }
  mx = red[0]; __syncthreads();

  float sum = 0.f;
  for (int j = tid; j < N; j += 256) {
    const float val = eb[(long long)i * N + j] + eb[(long long)j * N + i];
    const float a = ab[(long long)i * N + j];
    sum += __expf(((a > 0.f) ? val : -9e15f) - mx);
  }
  red[tid] = sum; __syncthreads();
  for (int s = 128; s > 0; s >>= 1) {
    if (tid < s) red[tid] += red[tid + s];
    __syncthreads();
  }
  const float inv = 1.f / red[0];

  for (int j = tid; j < N; j += 256) {
    const float val = eb[(long long)i * N + j] + eb[(long long)j * N + i];
    const float a = ab[(long long)i * N + j];
    ob[(long long)i * N + j] = __expf(((a > 0.f) ? val : -9e15f) - mx) * inv * a;
  }
}

// ---------------------------------------------------------------------------
// Gate: coeff = sigmoid([x,hp]@gw + gb); x = coeff*x + (1-coeff)*hp (in place)
// One wave per row (D=128 -> 4 elems/lane). grid = rows/4, block = 128.
// ---------------------------------------------------------------------------
__global__ void gate_kernel(float* __restrict__ x, const float* __restrict__ hp,
                            const float* __restrict__ gw,
                            const float* __restrict__ gb, int rows) {
  const int lane = threadIdx.x & 31;
  const int row = blockIdx.x * 4 + (threadIdx.x >> 5);
  if (row >= rows) return;
  float* xr = x + (long long)row * DD;
  const float* hr = hp + (long long)row * DD;
  float xs[4], hs[4], p = 0.f;
#pragma unroll
  for (int t = 0; t < 4; ++t) {
    const int d = lane + 32 * t;
    xs[t] = xr[d]; hs[t] = hr[d];
    p += xs[t] * gw[d] + hs[t] * gw[DD + d];
  }
#pragma unroll
  for (int off = 16; off > 0; off >>= 1) p += __shfl_xor(p, off, 32);
  const float c = sigmoidf_(p + gb[0]);
#pragma unroll
  for (int t = 0; t < 4; ++t) {
    const int d = lane + 32 * t;
    xr[d] = c * xs[t] + (1.f - c) * hs[t];
  }
}

// ---------------------------------------------------------------------------
// dm / adj12 from displacement vectors
// ---------------------------------------------------------------------------
__global__ void dm_kernel(const float* __restrict__ dmv, float* __restrict__ dm,
                          float* __restrict__ adj12, int total) {
  const int t = blockIdx.x * blockDim.x + threadIdx.x;
  if (t >= total) return;
  const float x = dmv[3 * t], y = dmv[3 * t + 1], z = dmv[3 * t + 2];
  const float r = sqrtf(x * x + y * y + z * z + 1e-10f);
  adj12[t] = (r > 0.001f && r <= 5.f) ? 1.f : 0.f;
  dm[t] = (r < 0.5f) ? 1e10f : r;
}

// ---------------------------------------------------------------------------
// EdgeConv message: m[b,a,d] = sum_c relu(dmv[b,pair]·cw[:,d]+cb[d]) * x2[b,c,d] * adj
// pair index = a*sa + c*sc over the [N1,N2] grid. grid=(Na,B), block=128 (=D).
// dmv/adj rows staged into LDS with gfx1250 async loads.
// ---------------------------------------------------------------------------
__global__ void edge_msg_kernel(const float* __restrict__ dmv,
                                const float* __restrict__ adj12,
                                const float* __restrict__ x2,
                                const float* __restrict__ cw,
                                const float* __restrict__ cb,
                                float* __restrict__ m,
                                int Na, int Nc, int sa, int sc) {
  const int b = blockIdx.y, a = blockIdx.x, d = threadIdx.x;
  __shared__ float s_dmv[128 * 3];
  __shared__ float s_adj[128];
  const float c0 = cw[d], c1 = cw[DD + d], c2 = cw[2 * DD + d], cbd = cb[d];
  float acc = 0.f;
  const long long pbase = (long long)b * N1_ * N2_;

  for (int cb0 = 0; cb0 < Nc; cb0 += 128) {
    const long long pidx = pbase + (long long)a * sa + (long long)(cb0 + d) * sc;
    async_copy_b32(dmv + 3 * pidx + 0, &s_dmv[3 * d + 0]);
    async_copy_b32(dmv + 3 * pidx + 1, &s_dmv[3 * d + 1]);
    async_copy_b32(dmv + 3 * pidx + 2, &s_dmv[3 * d + 2]);
    async_copy_b32(adj12 + pidx, &s_adj[d]);
    wait_async0();
    __syncthreads();

    const float* x2p = x2 + ((long long)b * Nc + cb0) * DD + d;
    if (cb0 + 128 < Nc)
      __builtin_prefetch(x2p + 128LL * DD, 0, 1);
#pragma unroll 4
    for (int cc = 0; cc < 128; ++cc) {
      const float av = s_adj[cc];  // uniform across the block
      if (av != 0.f) {
        float e = s_dmv[3 * cc] * c0 + s_dmv[3 * cc + 1] * c1 +
                  s_dmv[3 * cc + 2] * c2 + cbd;
        e = fmaxf(e, 0.f);
        acc += e * x2p[(long long)cc * DD];
      }
    }
    __syncthreads();
  }
  m[((long long)b * Na + a) * DD + d] = acc;
}

// ---------------------------------------------------------------------------
// Fused pair-MLP heads + Morse potential + reduction to out[b]
// p1X already contains +b1. grid = (N1, B), block = 128.
// ---------------------------------------------------------------------------
__global__ void pair_morse_kernel(const float* __restrict__ p1A, const float* __restrict__ p1B,
                                  const float* __restrict__ p1C, const float* __restrict__ p2A,
                                  const float* __restrict__ p2B, const float* __restrict__ p2C,
                                  const float* __restrict__ w2A, const float* __restrict__ w2B,
                                  const float* __restrict__ w2C, const float* __restrict__ b2A,
                                  const float* __restrict__ b2B, const float* __restrict__ b2C,
                                  const float* __restrict__ dm, float* __restrict__ out) {
  const int b = blockIdx.y, i = blockIdx.x, tid = threadIdx.x;
  __shared__ float sp[3][HID_];
  __shared__ float sw[3][HID_];
  __shared__ float red[128];
  const long long r1 = (long long)(b * N1_ + i) * HID_;
  sp[0][tid] = p1A[r1 + tid]; sp[1][tid] = p1B[r1 + tid]; sp[2][tid] = p1C[r1 + tid];
  sw[0][tid] = w2A[tid];      sw[1][tid] = w2B[tid];      sw[2][tid] = w2C[tid];
  __syncthreads();
  const float bA = b2A[0], bB = b2B[0], bC = b2C[0];

  float local = 0.f;
  for (int j = tid; j < N2_; j += 128) {
    const float* qA = p2A + (long long)(b * N2_ + j) * HID_;
    const float* qB = p2B + (long long)(b * N2_ + j) * HID_;
    const float* qC = p2C + (long long)(b * N2_ + j) * HID_;
    float aA = 0.f, aB = 0.f, aC = 0.f;
#pragma unroll 4
    for (int h = 0; h < HID_; ++h) {
      aA += fmaxf(sp[0][h] + qA[h], 0.f) * sw[0][h];
      aB += fmaxf(sp[1][h] + qB[h], 0.f) * sw[1][h];
      aC += fmaxf(sp[2][h] + qC[h], 0.f) * sw[2][h];
    }
    const float Av = 4.f * sigmoidf_(aA + bA);
    const float Bv = 4.f * sigmoidf_(aB + bB);
    const float Cv = 4.f * sigmoidf_(aC + bC);
    const float r = dm[((long long)b * N1_ + i) * N2_ + j];
    const float t = 1.f - __expf(-Bv * (r - Cv));
    local += Av * (t * t - 1.f);
  }
  red[tid] = local; __syncthreads();
  for (int s = 64; s > 0; s >>= 1) {
    if (tid < s) red[tid] += red[tid + s];
    __syncthreads();
  }
  if (tid == 0) atomicAdd(out + b, red[0]);
}

// ---------------------------------------------------------------------------
// Host orchestration
// ---------------------------------------------------------------------------
extern "C" void kernel_launch(void* const* d_in, const int* in_sizes, int n_in,
                              void* d_out, int out_size, void* d_ws, size_t ws_size,
                              hipStream_t stream) {
  const float* h1_in = (const float*)d_in[0];
  const float* adj1  = (const float*)d_in[1];
  const float* h2_in = (const float*)d_in[2];
  const float* adj2  = (const float*)d_in[3];
  const float* dmv   = (const float*)d_in[4];
  // d_in[5] = valid (unused by reference math)
  const float* W_emb  = (const float*)d_in[6];
  const float* gat_w  = (const float*)d_in[7];
  const float* gat_b  = (const float*)d_in[8];
  const float* gat_A  = (const float*)d_in[9];
  const float* gat_gw = (const float*)d_in[10];
  const float* gat_gb = (const float*)d_in[11];
  const float* ec_cw  = (const float*)d_in[12];
  const float* ec_cb  = (const float*)d_in[13];
  const float* ec_ww  = (const float*)d_in[14];
  const float* ec_wb  = (const float*)d_in[15];
  const float* ec_gw  = (const float*)d_in[16];
  const float* ec_gb  = (const float*)d_in[17];
  const float* cal_w1[3] = {(const float*)d_in[18], (const float*)d_in[22], (const float*)d_in[26]};
  const float* cal_b1[3] = {(const float*)d_in[19], (const float*)d_in[23], (const float*)d_in[27]};
  const float* cal_w2[3] = {(const float*)d_in[20], (const float*)d_in[24], (const float*)d_in[28]};
  const float* cal_b2[3] = {(const float*)d_in[21], (const float*)d_in[25], (const float*)d_in[29]};

  // ---- workspace layout (persistent + phase-overlaid scratch) ----
  char* ws = (char*)d_ws;
  size_t off = 0;
  auto alloc = [&](size_t nfloats) -> float* {
    float* p = (float*)(ws + off);
    off += ((nfloats * sizeof(float) + 255) / 256) * 256;
    return p;
  };
  float* h1    = alloc((size_t)BB * N1_ * DD);
  float* h2    = alloc((size_t)BB * N2_ * DD);
  float* dm    = alloc((size_t)BB * N1_ * N2_);
  float* adj12 = alloc((size_t)BB * N1_ * N2_);
  // transposed weights (persistent: alive across all phases)
  float* WpT     = alloc((size_t)DD * FEATP);
  float* gat_wT  = alloc((size_t)LL * DD * DD);
  float* gat_AT  = alloc((size_t)LL * DD * DD);
  float* ec_wwT  = alloc((size_t)LL * 2 * DD * DD);   // per layer: [loT | hiT]
  float* cal_w1T = alloc((size_t)3 * 2 * DD * DD);    // per head:  [loT | hiT]
  const size_t scratch0 = off;

  // Phase 0 (embedding) scratch: K-padded inputs and weight
  off = scratch0;
  float* hp1 = alloc((size_t)BB * N1_ * FEATP);
  float* hp2 = alloc((size_t)BB * N2_ * FEATP);
  float* Wp  = alloc((size_t)FEATP * DD);

  // Phase 1 (GAT) scratch — overlays phase 0
  off = scratch0;
  float* t_h   = alloc((size_t)BB * N2_ * DD);
  float* t_hT  = alloc((size_t)BB * N2_ * DD);
  float* t_hA  = alloc((size_t)BB * N2_ * DD);
  float* hp    = alloc((size_t)BB * N2_ * DD);
  float* e_buf = alloc((size_t)BB * N2_ * N2_);
  float* att   = alloc((size_t)BB * N2_ * N2_);

  // Phase 2 (EdgeConv) scratch — overlays phase 0/1
  off = scratch0;
  float* m1   = alloc((size_t)BB * N1_ * DD);
  float* m2   = alloc((size_t)BB * N2_ * DD);
  float* hec1 = alloc((size_t)BB * N1_ * DD);
  float* hec2 = alloc((size_t)BB * N2_ * DD);

  // Phase 3 (pair MLP) scratch — overlays phase 0/1/2
  off = scratch0;
  float* p1[3]; float* p2[3];
  for (int x = 0; x < 3; ++x) {
    p1[x] = alloc((size_t)BB * N1_ * HID_);
    p2[x] = alloc((size_t)BB * N2_ * HID_);
  }

  auto gemm = [&](const float* A, const float* Bt, const float* bias, float* C,
                  int M, int N, int K, int batch,
                  long long sA, long long sB, long long sC,
                  int accum, int act) {
    dim3 g((N + 63) / 64, (M + 63) / 64, batch);
    wmma_gemm_kernel<<<g, dim3(128), 0, stream>>>(A, Bt, bias, C, M, N, K,
                                                  sA, sB, sC, accum, act);
  };
  auto transpose = [&](const float* src, float* dst, int R, int C, int batch,
                       long long sS, long long sD) {
    transpose16_kernel<<<dim3(C / 16, R / 16, batch), 256, 0, stream>>>(
        src, dst, R, C, sS, sD);
  };

  // ---- weight prep: pad + transpose every B operand to [N,K] layout ----
  pad2d_kernel<<<(FEATP * DD + 255) / 256, 256, 0, stream>>>(
      W_emb, Wp, FEATK, DD, FEATP, DD);
  transpose(Wp, WpT, FEATP, DD, 1, 0, 0);
  for (int l = 0; l < LL; ++l) {
    transpose(gat_w + (size_t)l * DD * DD, gat_wT + (size_t)l * DD * DD, DD, DD, 1, 0, 0);
    transpose(gat_A + (size_t)l * DD * DD, gat_AT + (size_t)l * DD * DD, DD, DD, 1, 0, 0);
    const float* lo = ec_ww + (size_t)l * 2 * DD * DD;
    transpose(lo,            ec_wwT + (size_t)l * 2 * DD * DD,            DD, DD, 1, 0, 0);
    transpose(lo + DD * DD,  ec_wwT + (size_t)l * 2 * DD * DD + DD * DD,  DD, DD, 1, 0, 0);
  }
  for (int x = 0; x < 3; ++x) {
    transpose(cal_w1[x],           cal_w1T + (size_t)x * 2 * DD * DD,           DD, DD, 1, 0, 0);
    transpose(cal_w1[x] + DD * DD, cal_w1T + (size_t)x * 2 * DD * DD + DD * DD, DD, DD, 1, 0, 0);
  }

  // ---- embedding (K padded 56 -> 64; no bounds checks in the GEMM) ----
  pad2d_kernel<<<(BB * N1_ * FEATP + 255) / 256, 256, 0, stream>>>(
      h1_in, hp1, BB * N1_, FEATK, BB * N1_, FEATP);
  pad2d_kernel<<<(BB * N2_ * FEATP + 255) / 256, 256, 0, stream>>>(
      h2_in, hp2, BB * N2_, FEATK, BB * N2_, FEATP);
  gemm(hp1, WpT, nullptr, h1, BB * N1_, DD, FEATP, 1, 0, 0, 0, 0, ACT_NONE);
  gemm(hp2, WpT, nullptr, h2, BB * N2_, DD, FEATP, 1, 0, 0, 0, 0, ACT_NONE);

  // ---- GAT layers (shared weights across the two graphs) ----
  auto run_gat = [&](float* x, const float* adj, int N, int l) {
    const float* wT = gat_wT + (size_t)l * DD * DD;
    const float* bb = gat_b + (size_t)l * DD;
    const float* AT = gat_AT + (size_t)l * DD * DD;
    const float* gw = gat_gw + (size_t)l * 2 * DD;
    const float* gb = gat_gb + l;
    gemm(x, wT, bb, t_h, BB * N, DD, DD, 1, 0, 0, 0, 0, ACT_NONE);
    gemm(t_h, AT, nullptr, t_hA, BB * N, DD, DD, 1, 0, 0, 0, 0, ACT_NONE);
    // e[b] = (h@A)[b] @ h[b]^T : Bt = t_h itself ([N,K=DD] rows)
    gemm(t_hA, t_h, nullptr, e_buf, N, N, DD, BB,
         (long long)N * DD, (long long)N * DD, (long long)N * N, 0, ACT_NONE);
    att_softmax_kernel<<<dim3(N, BB), 256, 0, stream>>>(e_buf, adj, att, N);
    // hp[b] = relu(att[b] @ h[b]) : Bt = transpose(t_h[b]) [DD, N]
    transpose(t_h, t_hT, N, DD, BB, (long long)N * DD, (long long)N * DD);
    gemm(att, t_hT, nullptr, hp, N, DD, N, BB,
         (long long)N * N, (long long)N * DD, (long long)N * DD, 0, ACT_RELU);
    gate_kernel<<<dim3((BB * N) / 4), 128, 0, stream>>>(x, hp, gw, gb, BB * N);
  };
  for (int l = 0; l < LL; ++l) {
    run_gat(h1, adj1, N1_, l);
    run_gat(h2, adj2, N2_, l);
  }

  // ---- distance map / cross adjacency ----
  {
    const int total = BB * N1_ * N2_;
    dm_kernel<<<(total + 255) / 256, 256, 0, stream>>>(dmv, dm, adj12, total);
  }

  // ---- EdgeConv layers ----
  for (int l = 0; l < LL; ++l) {
    const float* cw    = ec_cw + (size_t)l * 3 * DD;
    const float* cbp   = ec_cb + (size_t)l * DD;
    const float* wwloT = ec_wwT + (size_t)l * 2 * DD * DD;
    const float* wwhiT = wwloT + (size_t)DD * DD;
    const float* wbp   = ec_wb + (size_t)l * DD;
    const float* gw    = ec_gw + (size_t)l * 2 * DD;
    const float* gb    = ec_gb + l;

    edge_msg_kernel<<<dim3(N1_, BB), 128, 0, stream>>>(dmv, adj12, h2, cw, cbp,
                                                       m1, N1_, N2_, N2_, 1);
    edge_msg_kernel<<<dim3(N2_, BB), 128, 0, stream>>>(dmv, adj12, h1, cw, cbp,
                                                       m2, N2_, N1_, 1, N2_);
    // h = relu(x1@ww_lo + m@ww_hi + wb)
    gemm(h1, wwloT, wbp, hec1, BB * N1_, DD, DD, 1, 0, 0, 0, 0, ACT_NONE);
    gemm(m1, wwhiT, nullptr, hec1, BB * N1_, DD, DD, 1, 0, 0, 0, 1, ACT_RELU);
    gemm(h2, wwloT, wbp, hec2, BB * N2_, DD, DD, 1, 0, 0, 0, 0, ACT_NONE);
    gemm(m2, wwhiT, nullptr, hec2, BB * N2_, DD, DD, 1, 0, 0, 0, 1, ACT_RELU);
    // gated residual, in place (all readers of old h1/h2 already issued)
    gate_kernel<<<dim3((BB * N1_) / 4), 128, 0, stream>>>(h1, hec1, gw, gb, BB * N1_);
    gate_kernel<<<dim3((BB * N2_) / 4), 128, 0, stream>>>(h2, hec2, gw, gb, BB * N2_);
  }

  // ---- pair-MLP projections (bias b1 folded into p1) ----
  for (int x = 0; x < 3; ++x) {
    const float* loT = cal_w1T + (size_t)x * 2 * DD * DD;
    const float* hiT = loT + (size_t)DD * DD;
    gemm(h1, loT, cal_b1[x], p1[x], BB * N1_, HID_, DD, 1, 0, 0, 0, 0, ACT_NONE);
    gemm(h2, hiT, nullptr,   p2[x], BB * N2_, HID_, DD, 1, 0, 0, 0, 0, ACT_NONE);
  }

  // ---- fused Morse potential + per-batch reduction ----
  (void)hipMemsetAsync(d_out, 0, BB * sizeof(float), stream);
  pair_morse_kernel<<<dim3(N1_, BB), 128, 0, stream>>>(
      p1[0], p1[1], p1[2], p2[0], p2[1], p2[2],
      cal_w2[0], cal_w2[1], cal_w2[2],
      cal_b2[0], cal_b2[1], cal_b2[2],
      dm, (float*)d_out);
}
